// DecoderRNN_13426067767752
// MI455X (gfx1250) — matrix operand order repaired and checked by reading, not compiled, not run on hardware
//
#include <hip/hip_runtime.h>

typedef __bf16 bf16;
typedef __attribute__((ext_vector_type(16))) __bf16 v16bf;
typedef __attribute__((ext_vector_type(8)))  __bf16 v8bf;
typedef __attribute__((ext_vector_type(8)))  float  v8f;
typedef __attribute__((ext_vector_type(4)))  unsigned int v4u;
typedef __attribute__((ext_vector_type(8)))  unsigned int v8u;
typedef __attribute__((ext_vector_type(4)))  unsigned int u32x4;
typedef __attribute__((ext_vector_type(4)))  int i32x4;
typedef __attribute__((ext_vector_type(8)))  int i32x8;

#define V_DIM 32000
#define E_DIM 512
#define H_DIM 512
#define L_DIM 2
#define B_DIM 16
#define S_DIM 128
#define SE_DIM 256

__device__ __forceinline__ float sigm(float x) { return 1.0f / (1.0f + __expf(-x)); }

__device__ __forceinline__ v8f wmma_bf16(v16bf a, v16bf b, v8f c) {
  return __builtin_amdgcn_wmma_f32_16x16x32_bf16(false, a, false, b, (short)0, c, false, false);
}

// A-matrix fragment (16x32, M rows). Lane<16: row=lane, K=kb+0..15.
// Lane>=16: row=lane-16, K=kb+16..31.  (two 16B loads)
__device__ __forceinline__ v16bf load_afrag(const bf16* base, int ld, int kb, int lane) {
  const bf16* p = base + (size_t)(lane & 15) * ld + kb + ((lane >> 4) << 4);
  v4u lo = *(const v4u*)p;
  v4u hi = *(const v4u*)(p + 8);
  v8u c = __builtin_shufflevector(lo, hi, 0, 1, 2, 3, 4, 5, 6, 7);
  return __builtin_bit_cast(v16bf, c);
}

// B-matrix fragment (32x16, KxN) from row-major W[N][K]: 32 contiguous bytes per lane.
__device__ __forceinline__ v16bf load_bfrag(const bf16* W, int ld, int nrow, int kb, int lane) {
  const bf16* p = W + (size_t)(nrow + (lane & 15)) * ld + kb + ((lane >> 4) << 4);
  v8u c = *(const v8u*)p;
  return __builtin_bit_cast(v16bf, c);
}

// ---------------------------------------------------------------------------
// Tensor Data Mover: DMA a 2D tile [rows=128][cols=32 bf16] from row-major
// global (row stride 512 elems) into LDS at byte offset lds_off.
// D# layout per CDNA5 ISA ch.8 (group0 128b, group1 256b; groups 2/3 unused).
// ---------------------------------------------------------------------------
__device__ __forceinline__ void tdm_load_tile_128x32(unsigned lds_off, const bf16* gaddr) {
  unsigned long long ga = (unsigned long long)(const void*)gaddr;
  u32x4 g0;
  g0[0] = 1u;                                            // count=1, user descriptor
  g0[1] = lds_off;                                       // lds_addr (bytes)
  g0[2] = (unsigned)(ga & 0xffffffffu);                  // global_addr[31:0]
  g0[3] = (unsigned)((ga >> 32) & 0x01ffffffu) | 0x80000000u;  // addr[56:32] | type=2
  i32x8 g1;
  g1[0] = 0x00010000;            // data_size=1 (2 bytes/elem), no multicast/pad/iterate
  g1[1] = (int)(512u << 16);     // tensor_dim0 = 512 (bits 79:48, lo16)
  g1[2] = (int)(0x7D00u << 16);  // tensor_dim0 hi16=0 | tensor_dim1 = 32000 (lo16)
  g1[3] = (int)(32u << 16);      // tensor_dim1 hi16=0 | tile_dim0 = 32
  g1[4] = 128;                   // tile_dim1 = 128, tile_dim2 = 0
  g1[5] = 512;                   // tensor_dim0_stride lo32 = 512
  g1[6] = 0;                     // stride hi16 = 0 | tensor_dim1_stride lo16 = 0
  g1[7] = 0;
  i32x4 gz = {0, 0, 0, 0};
#if __clang_major__ >= 23
  i32x8 gz8 = {0, 0, 0, 0, 0, 0, 0, 0};
  __builtin_amdgcn_tensor_load_to_lds(g0, g1, gz, gz, gz8, 0);
#else
  __builtin_amdgcn_tensor_load_to_lds(g0, g1, gz, gz, 0);
#endif
}

// ---------------------------------------------------------------------------
// prep kernels
// ---------------------------------------------------------------------------
__global__ void cast_bf16_kernel(const float* __restrict__ s, bf16* __restrict__ d, int n) {
  int i = blockIdx.x * blockDim.x + threadIdx.x;
  int stride = gridDim.x * blockDim.x;
  for (; i < n; i += stride) d[i] = (bf16)s[i];
}

__global__ void embed_kernel(const int* __restrict__ tok, const float* __restrict__ emb,
                             bf16* __restrict__ x_all) {  // x_all: [S][B][E]
  int idx = blockIdx.x * blockDim.x + threadIdx.x;
  if (idx >= S_DIM * B_DIM * E_DIM) return;
  int e = idx & (E_DIM - 1);
  int b = (idx >> 9) & (B_DIM - 1);
  int s = idx >> 13;
  int t = tok[b * S_DIM + s];
  x_all[idx] = (bf16)emb[(size_t)t * E_DIM + e];
}

// ---------------------------------------------------------------------------
// sequential GRU + attention scan: ONE persistent workgroup (32 wave32 waves)
// ---------------------------------------------------------------------------
__global__ __launch_bounds__(1024) void scan_kernel(
    const bf16* __restrict__ x_all,          // [S][B][E] bf16
    const bf16* __restrict__ wih,            // [L][3H][H] bf16
    const bf16* __restrict__ whh,            // [L][3H][H] bf16
    const float* __restrict__ b_ih,          // [L][3H]
    const float* __restrict__ b_hh,          // [L][3H]
    const bf16* __restrict__ wain,           // [H][H] bf16
    const bf16* __restrict__ waout,          // [H][2H] bf16
    const bf16* __restrict__ enc,            // [B][SE][H] bf16
    const unsigned char* __restrict__ mask,  // [B][SE]
    const float* __restrict__ h0,            // [L][B][H] f32
    bf16* __restrict__ outs,                 // [B][S][H] bf16 (tanh'd)
    float* __restrict__ hfin,                // [L][B][H] f32
    float* __restrict__ attn_out)            // [S][B][SE] f32
{
  __shared__ __align__(16) bf16 hbf[2 * 16 * 512];   // 32 KB  bf16 mirror of h (both layers)
  __shared__ __align__(16) bf16 shx[16 * 512];       // 16 KB  x -> q -> mix (time-multiplexed)
  __shared__ __align__(16) bf16 attnb[16 * 256];     //  8 KB
  __shared__ float red[16 * 64];                     //  4 KB  softmax reductions

  const int tid = threadIdx.x;
  const int lane = tid & 31;
  const int wave = tid >> 5;       // 0..31
  const int lh = lane >> 4;        // 0/1
  const int n15 = lane & 15;
  const int col = wave * 16 + n15; // this wave's h-column for this lane

  // persistent hidden state in WMMA C-fragment layout: hreg[l][j] = h[l][m][col], m=j+8*lh
  float hreg[2][8];
  for (int l = 0; l < 2; ++l)
    for (int j = 0; j < 8; ++j) {
      int m = j + (lh << 3);
      float v = h0[(l * 16 + m) * 512 + col];
      hreg[l][j] = v;
      hbf[(l * 16 + m) * 512 + col] = (bf16)v;
    }
  float biasR[2], biasZ[2], biasXN[2], biasHN[2];
  for (int l = 0; l < 2; ++l) {
    biasR[l]  = b_ih[l * 1536 + col]        + b_hh[l * 1536 + col];
    biasZ[l]  = b_ih[l * 1536 + 512 + col]  + b_hh[l * 1536 + 512 + col];
    biasXN[l] = b_ih[l * 1536 + 1024 + col];
    biasHN[l] = b_hh[l * 1536 + 1024 + col];
  }
  __syncthreads();

  for (int s = 0; s < S_DIM; ++s) {
    // stage this step's embedded input [16][512] bf16 into LDS
    ((v4u*)shx)[tid] = ((const v4u*)(x_all + (size_t)s * 8192))[tid];
    __syncthreads();

    // ---- 2 GRU layers: wave w owns h-slice [16w,16w+16); gate tiles r(w), z(32+w), n(64+w).
    // Two passes (x-side then h-side) to keep only 3 live accumulators per pass.
    for (int l = 0; l < 2; ++l) {
      const bf16* xsrc = (l == 0) ? shx : hbf;  // layer1 input = layer0's new h
      const bf16* hsrc = hbf + l * 8192;
      const bf16* Wx = wih + (size_t)l * 1536 * 512;
      const bf16* Wh = whh + (size_t)l * 1536 * 512;
      v8f cxr = {0,0,0,0,0,0,0,0}, cxz = cxr, cxn = cxr;
#pragma unroll 4
      for (int kc = 0; kc < 16; ++kc) {
        const int kb = kc * 32;
        v16bf a  = load_afrag(xsrc, 512, kb, lane);
        v16bf br = load_bfrag(Wx, 512, wave * 16,        kb, lane);
        v16bf bz = load_bfrag(Wx, 512, 512  + wave * 16, kb, lane);
        v16bf bn = load_bfrag(Wx, 512, 1024 + wave * 16, kb, lane);
        cxr = wmma_bf16(a, br, cxr);
        cxz = wmma_bf16(a, bz, cxz);
        cxn = wmma_bf16(a, bn, cxn);
      }
      v8f chr_ = {0,0,0,0,0,0,0,0}, chz = chr_, chn = chr_;
#pragma unroll 4
      for (int kc = 0; kc < 16; ++kc) {
        const int kb = kc * 32;
        v16bf a  = load_afrag(hsrc, 512, kb, lane);
        v16bf br = load_bfrag(Wh, 512, wave * 16,        kb, lane);
        v16bf bz = load_bfrag(Wh, 512, 512  + wave * 16, kb, lane);
        v16bf bn = load_bfrag(Wh, 512, 1024 + wave * 16, kb, lane);
        chr_ = wmma_bf16(a, br, chr_);
        chz  = wmma_bf16(a, bz, chz);
        chn  = wmma_bf16(a, bn, chn);
      }
      __syncthreads();  // all waves done READING hbf before overwriting it
      for (int j = 0; j < 8; ++j) {
        float r  = sigm(cxr[j] + chr_[j] + biasR[l]);
        float z  = sigm(cxz[j] + chz[j] + biasZ[l]);
        float nn = tanhf(cxn[j] + biasXN[l] + r * (chn[j] + biasHN[l]));
        float hp = (1.0f - z) * nn + z * hreg[l][j];
        hreg[l][j] = hp;
        int m = j + (lh << 3);
        hbf[(l * 16 + m) * 512 + col] = (bf16)hp;
      }
      __syncthreads();
    }

    // ---- q = h_top @ w_attn_in^T  (WMMA, one N-tile per wave) -> shx as bf16
    {
      v8f q = {0,0,0,0,0,0,0,0};
#pragma unroll 4
      for (int kc = 0; kc < 16; ++kc) {
        v16bf a = load_afrag(hbf + 8192, 512, kc * 32, lane);
        q = wmma_bf16(a, load_bfrag(wain, 512, wave * 16, kc * 32, lane), q);
      }
      for (int j = 0; j < 8; ++j) {
        int m = j + (lh << 3);
        shx[m * 512 + wave * 16 + n15] = (bf16)q[j];
      }
    }
    __syncthreads();

    // ---- scores + masked softmax (per-batch B operand -> packed VALU).
    {
      const int b = tid >> 6;
      const int si = tid & 63;
      float sc[4];
      for (int i = 0; i < 4; ++i) {
        const int ss = si * 4 + i;
        v8f acc8 = {0,0,0,0,0,0,0,0};
        const bf16* ep = enc + ((size_t)(b * 256 + ss)) * 512;
        const bf16* qp = shx + b * 512;
        for (int h = 0; h < 512; h += 8) {
          v8bf e8 = __builtin_bit_cast(v8bf, *(const v4u*)(ep + h));
          v8bf q8 = __builtin_bit_cast(v8bf, *(const v4u*)(qp + h));
          acc8 += __builtin_convertvector(e8, v8f) * __builtin_convertvector(q8, v8f);
        }
        float acc = ((acc8[0] + acc8[4]) + (acc8[1] + acc8[5])) +
                    ((acc8[2] + acc8[6]) + (acc8[3] + acc8[7]));
        sc[i] = mask[b * 256 + ss] ? acc : -1e20f;
      }
      float pm = fmaxf(fmaxf(sc[0], sc[1]), fmaxf(sc[2], sc[3]));
      red[b * 64 + si] = pm;
      __syncthreads();
      float mx = -3.4e38f;
      for (int k = 0; k < 64; ++k) mx = fmaxf(mx, red[b * 64 + k]);
      __syncthreads();
      float ps = 0.0f;
      for (int i = 0; i < 4; ++i) { sc[i] = __expf(sc[i] - mx); ps += sc[i]; }
      red[b * 64 + si] = ps;
      __syncthreads();
      float sum = 0.0f;
      for (int k = 0; k < 64; ++k) sum += red[b * 64 + k];
      const float inv = 1.0f / sum;
      for (int i = 0; i < 4; ++i) {
        const int ss = si * 4 + i;
        const float a = sc[i] * inv;
        attnb[b * 256 + ss] = (bf16)a;
        attn_out[((size_t)s * 16 + b) * 256 + ss] = a;
      }
    }
    __syncthreads();

    // ---- mix = attn @ enc (packed VALU) -> shx (overwrites q; q dead now)
    {
      const int b = tid >> 6;
      const int hb = (tid & 63) * 8;
      v8f acc = {0,0,0,0,0,0,0,0};
      for (int ss = 0; ss < 256; ++ss) {
        const float a = (float)attnb[b * 256 + ss];
        v8bf e8 = __builtin_bit_cast(v8bf, *(const v4u*)(enc + ((size_t)(b * 256 + ss)) * 512 + hb));
        acc += __builtin_convertvector(e8, v8f) * a;
      }
      v8bf o = __builtin_convertvector(acc, v8bf);
      *(v4u*)(shx + b * 512 + hb) = __builtin_bit_cast(v4u, o);
    }
    __syncthreads();

    // ---- out = tanh(concat(h_top, mix) @ w_attn_out^T)  (WMMA, one N-tile per wave)
    {
      v8f c = {0,0,0,0,0,0,0,0};
#pragma unroll 4
      for (int kc = 0; kc < 16; ++kc) {  // h_top half (k = 0..511)
        v16bf a = load_afrag(hbf + 8192, 512, kc * 32, lane);
        c = wmma_bf16(a, load_bfrag(waout, 1024, wave * 16, kc * 32, lane), c);
      }
#pragma unroll 4
      for (int kc = 0; kc < 16; ++kc) {  // mix half (k = 512..1023)
        v16bf a = load_afrag(shx, 512, kc * 32, lane);
        c = wmma_bf16(a, load_bfrag(waout, 1024, wave * 16, 512 + kc * 32, lane), c);
      }
      for (int j = 0; j < 8; ++j) {
        int m = j + (lh << 3);
        outs[((size_t)m * S_DIM + s) * 512 + wave * 16 + n15] = (bf16)tanhf(c[j]);
      }
    }
    __syncthreads();  // before next step's x overwrites shx
  }

  for (int l = 0; l < 2; ++l)
    for (int j = 0; j < 8; ++j) {
      int m = j + (lh << 3);
      hfin[(l * 16 + m) * 512 + col] = hreg[l][j];
    }
}

// ---------------------------------------------------------------------------
// logits GEMM: [2048 x 512] x [512 x 32000] + bias, bf16 WMMA, f32 out.
// block = 256 threads (8 waves); tile 128(M) x 128(N).
// B tiles staged into LDS by the Tensor Data Mover, double-buffered.
// ---------------------------------------------------------------------------
__global__ __launch_bounds__(256) void logits_kernel(
    const bf16* __restrict__ A,     // outs_bf [2048][512]
    const bf16* __restrict__ W,     // w_out_bf [32000][512]
    const float* __restrict__ bias, // [32000]
    float* __restrict__ out)        // [2048][32000]
{
  __shared__ __align__(16) bf16 btiles[2 * 128 * 32];  // 16 KB: two [n][k] k-chunk buffers
  const int tid = threadIdx.x;
  const int lane = tid & 31;
  const int wave = tid >> 5;                // 0..7 -> M-tile
  const int Nbase = blockIdx.x * 128;
  const int Mbase = blockIdx.y * 128;
  const bf16* Arow = A + (size_t)(Mbase + wave * 16) * 512;
  const bf16* Wblk = W + (size_t)Nbase * 512;
  const unsigned lds0 = (unsigned)(unsigned long long)(void*)btiles;  // LDS byte offset

  v8f acc[8];
  for (int nt = 0; nt < 8; ++nt) acc[nt] = (v8f){0,0,0,0,0,0,0,0};

  if (wave == 0) tdm_load_tile_128x32(lds0, Wblk);  // prime buffer 0 (kc=0)

  for (int kc = 0; kc < 16; ++kc) {
    if (wave == 0) {
      if (kc < 15) {
        tdm_load_tile_128x32(lds0 + ((kc + 1) & 1) * 8192, Wblk + (kc + 1) * 32);
        __builtin_amdgcn_s_wait_tensorcnt(1);  // oldest DMA (this kc's tile) done
      } else {
        __builtin_amdgcn_s_wait_tensorcnt(0);
      }
    }
    __syncthreads();  // tile kc visible to all waves
    const bf16* bt = btiles + (kc & 1) * (128 * 32);
    v16bf a = load_afrag(Arow, 512, kc * 32, lane);
    if (kc < 15) __builtin_prefetch(Arow + (size_t)(lane & 15) * 512 + (kc + 1) * 32, 0, 0);
#pragma unroll
    for (int nt = 0; nt < 8; ++nt) {
      const bf16* p = bt + (nt * 16 + (lane & 15)) * 32 + ((lane >> 4) << 4);
      v16bf b = __builtin_bit_cast(v16bf, *(const v8u*)p);
      acc[nt] = wmma_bf16(a, b, acc[nt]);
    }
    __syncthreads();  // reads done before TDM overwrites this buffer next-next iter
  }

  for (int nt = 0; nt < 8; ++nt) {
    const int cn = Nbase + nt * 16 + (lane & 15);
    const float bv = bias[cn];
#pragma unroll
    for (int j = 0; j < 8; ++j) {
      const int row = Mbase + wave * 16 + j + ((lane >> 4) << 3);
      out[(size_t)row * V_DIM + cn] = acc[nt][j] + bv;
    }
  }
}

// ---------------------------------------------------------------------------
extern "C" void kernel_launch(void* const* d_in, const int* in_sizes, int n_in,
                              void* d_out, int out_size, void* d_ws, size_t ws_size,
                              hipStream_t stream) {
  (void)in_sizes; (void)n_in; (void)out_size; (void)ws_size;
  const int*   text   = (const int*)  d_in[0];
  const float* dh     = (const float*)d_in[1];
  const float* encf   = (const float*)d_in[2];
  const unsigned char* msk = (const unsigned char*)d_in[3];
  const float* emb    = (const float*)d_in[4];
  const float* w_ih   = (const float*)d_in[5];
  const float* w_hh   = (const float*)d_in[6];
  const float* b_ih   = (const float*)d_in[7];
  const float* b_hh   = (const float*)d_in[8];
  const float* w_ain  = (const float*)d_in[9];
  const float* w_aout = (const float*)d_in[10];
  const float* w_out  = (const float*)d_in[11];
  const float* b_out  = (const float*)d_in[12];

  // workspace carve-up (bf16), ~47 MB total
  bf16* p = (bf16*)d_ws;
  bf16* wih_bf   = p;  p += 2 * 1536 * 512;
  bf16* whh_bf   = p;  p += 2 * 1536 * 512;
  bf16* wain_bf  = p;  p += 512 * 512;
  bf16* waout_bf = p;  p += 512 * 1024;
  bf16* wout_bf  = p;  p += (size_t)V_DIM * 512;
  bf16* enc_bf   = p;  p += 16 * 256 * 512;
  bf16* x_all    = p;  p += 128 * 16 * 512;
  bf16* outs_bf  = p;  p += 16 * 128 * 512;

  auto cast = [&](const float* s, bf16* d, int n) {
    cast_bf16_kernel<<<dim3(2048), dim3(256), 0, stream>>>(s, d, n);
  };
  cast(w_ih,   wih_bf,   2 * 1536 * 512);
  cast(w_hh,   whh_bf,   2 * 1536 * 512);
  cast(w_ain,  wain_bf,  512 * 512);
  cast(w_aout, waout_bf, 512 * 1024);
  cast(w_out,  wout_bf,  V_DIM * 512);
  cast(encf,   enc_bf,   16 * 256 * 512);

  embed_kernel<<<dim3((S_DIM * B_DIM * E_DIM + 255) / 256), dim3(256), 0, stream>>>(
      text, emb, x_all);

  float* logits  = (float*)d_out;                           // [B,S,V]
  float* hfin    = logits + (size_t)B_DIM * S_DIM * V_DIM;  // [L,B,H]
  float* attnlog = hfin + (size_t)L_DIM * B_DIM * H_DIM;    // [S,B,SE]

  scan_kernel<<<dim3(1), dim3(1024), 0, stream>>>(
      x_all, wih_bf, whh_bf, b_ih, b_hh, wain_bf, waout_bf, enc_bf, msk, dh,
      outs_bf, hfin, attnlog);

  logits_kernel<<<dim3(V_DIM / 128, (B_DIM * S_DIM) / 128), dim3(256), 0, stream>>>(
      outs_bf, wout_bf, b_out, logits);
}
